// Model_11355893530697
// MI455X (gfx1250) — matrix operand order repaired and checked by reading, not compiled, not run on hardware
//
#include <hip/hip_runtime.h>
#include <hip/hip_bf16.h>

// MI455X (gfx1250) implementation.
// - All dense GEMMs, attention score/PV products, and the LSTM recurrence use
//   v_wmma_f32_16x16x32_f16 (f16 operands, fp32 accumulate).
// - Attention stages each 128x128 V chunk into LDS via async global->LDS
//   (ASYNCcnt); PV fragments then come from LDS.
// - LSTM scan: Whh resident in LDS as f16 (128KB), h/c state in LDS, xg[t]
//   double-buffered in LDS and prefetched one step ahead with the Tensor Data
//   Mover (TENSORcnt).

typedef __attribute__((ext_vector_type(16))) _Float16 v16h;
typedef __attribute__((ext_vector_type(8)))  float    v8f;
typedef __attribute__((ext_vector_type(4)))  int      i4v;

#define WMMA_F16(Af, Bf, Cacc) \
  __builtin_amdgcn_wmma_f32_16x16x32_f16(false, (Af), false, (Bf), (short)0, (Cacc), false, false)

#if __has_builtin(__builtin_amdgcn_global_load_async_to_lds_b128)
#define HAVE_ASYNC_LDS 1
#endif

#if __has_builtin(__builtin_amdgcn_tensor_load_to_lds) && __has_builtin(__builtin_amdgcn_s_wait_tensorcnt)
#define HAVE_TDM 1
typedef __attribute__((ext_vector_type(4))) unsigned int tdm_v4u;
typedef __attribute__((ext_vector_type(8))) int          tdm_v8i;
typedef __attribute__((ext_vector_type(4))) int          tdm_v4i;
#endif

static __device__ __forceinline__ void wait_async0() {
#if defined(HAVE_ASYNC_LDS)
#if __has_builtin(__builtin_amdgcn_s_wait_asynccnt)
  __builtin_amdgcn_s_wait_asynccnt(0);
#else
  asm volatile("s_wait_asynccnt 0x0" ::: "memory");
#endif
#endif
}

static __device__ __forceinline__ float sigmoidf_(float x) {
  return 1.0f / (1.0f + __expf(-x));
}

// Load A fragment (16x32 f16, M x K) from row-major fp32 A, row = m0+idx.
static __device__ __forceinline__ v16h load_a_frag(const float* __restrict__ arow,
                                                   int k0, int half) {
  v16h af;
  const float4* p0 = (const float4*)(arow + k0 + half * 16);
  const float4* p1 = (const float4*)(arow + k0 + half * 16 + 8);
  float4 x0 = p0[0], x1 = p0[1], y0 = p1[0], y1 = p1[1];
  af[0]  = (_Float16)x0.x; af[1]  = (_Float16)x0.y; af[2]  = (_Float16)x0.z; af[3]  = (_Float16)x0.w;
  af[4]  = (_Float16)x1.x; af[5]  = (_Float16)x1.y; af[6]  = (_Float16)x1.z; af[7]  = (_Float16)x1.w;
  af[8]  = (_Float16)y0.x; af[9]  = (_Float16)y0.y; af[10] = (_Float16)y0.z; af[11] = (_Float16)y0.w;
  af[12] = (_Float16)y1.x; af[13] = (_Float16)y1.y; af[14] = (_Float16)y1.z; af[15] = (_Float16)y1.w;
  return af;
}

// Load B fragment (32x16 f16, K x N) where B[k][n] = W[n][k]; per-lane column
// n = n0+idx fixed, K contiguous: W[col][k0 + half*16 + i].
static __device__ __forceinline__ v16h load_b_frag(const float* __restrict__ wrow,
                                                   int k0, int half) {
  v16h bf;
  const float4* p = (const float4*)(wrow + k0 + half * 16);
  float4 w0 = p[0], w1 = p[1], w2 = p[2], w3 = p[3];
  bf[0]  = (_Float16)w0.x; bf[1]  = (_Float16)w0.y; bf[2]  = (_Float16)w0.z; bf[3]  = (_Float16)w0.w;
  bf[4]  = (_Float16)w1.x; bf[5]  = (_Float16)w1.y; bf[6]  = (_Float16)w1.z; bf[7]  = (_Float16)w1.w;
  bf[8]  = (_Float16)w2.x; bf[9]  = (_Float16)w2.y; bf[10] = (_Float16)w2.z; bf[11] = (_Float16)w2.w;
  bf[12] = (_Float16)w3.x; bf[13] = (_Float16)w3.y; bf[14] = (_Float16)w3.z; bf[15] = (_Float16)w3.w;
  return bf;
}

// ---------------------------------------------------------------------------
// Generic GEMM: C[M,N] = op(A[M,K] * W[N,K]^T + b1[n] + b2[n]); op = relu opt.
// One wave per 16x16 C tile; 8 waves (256 threads) per block.
// ---------------------------------------------------------------------------
__global__ __launch_bounds__(256) void gemm_wmma(
    const float* __restrict__ A, int lda,
    const float* __restrict__ W, int ldw,
    const float* __restrict__ b1, const float* __restrict__ b2,
    float* __restrict__ C, int ldc, int coloff,
    int M, int N, int K, int relu) {
  int wave = threadIdx.x >> 5;
  int lane = threadIdx.x & 31;
  int idx = lane & 15, half = lane >> 4;
  int nt = blockIdx.y * 8 + wave;
  if (nt * 16 >= N) return;
  int m0 = blockIdx.x * 16, n0 = nt * 16;
  const float* arow = A + (size_t)(m0 + idx) * lda;
  const float* wrow = W + (size_t)(n0 + idx) * ldw;
  v8f acc = {};
  for (int k0 = 0; k0 < K; k0 += 32) {
    v16h af = load_a_frag(arow, k0, half);
    v16h bf = load_b_frag(wrow, k0, half);
    acc = WMMA_F16(af, bf, acc);
  }
  int col = n0 + idx;
  float bias = (b1 ? b1[col] : 0.0f) + (b2 ? b2[col] : 0.0f);
#pragma unroll
  for (int r = 0; r < 8; ++r) {
    int row = m0 + half * 8 + r;
    float v = acc[r] + bias;
    if (relu) v = fmaxf(v, 0.0f);
    C[(size_t)row * ldc + coloff + col] = v;
  }
}

// ---------------------------------------------------------------------------
// Embedding gather: x[b,s,:] = emb[pid[b,s],:]
// ---------------------------------------------------------------------------
__global__ void gather_x(const int* __restrict__ pid, const float* __restrict__ emb,
                         float* __restrict__ X, int total) {
  int i = blockIdx.x * blockDim.x + threadIdx.x;
  if (i >= total) return;
  int row = i >> 6, d = i & 63;
  X[i] = emb[(size_t)pid[row] * 64 + d];
}

// ---------------------------------------------------------------------------
// Flash-style causal attention with dual qa-gated value aggregation.
// Writes all 4 slices of p_input[b,s,512] = [qa*v | (1-qa)*v | wv1 | wv0].
// Block: 256 threads (8 waves), one 16-query tile per block.
// Dynamic LDS: scores (8K) + P1/P0 f16 (8K) + stats + V chunk f32 (64K).
// ---------------------------------------------------------------------------
#define ATTN_SMEM_BYTES (16640 + 128 * 128 * 4)

__global__ __launch_bounds__(256) void attn_kernel(
    const float* __restrict__ Q, const float* __restrict__ Km,
    const float* __restrict__ V, const int* __restrict__ qa,
    float* __restrict__ PIN) {
  const int S = 1024, E = 128;
  extern __shared__ char smem[];
  float*    sS    = (float*)smem;                  // [16][128]
  _Float16* sP1   = (_Float16*)(smem + 8192);      // [16][128]
  _Float16* sP0   = (_Float16*)(smem + 12288);     // [16][128]
  float*    sMax  = (float*)(smem + 16384);        // [16]
  float*    sSum  = sMax + 16;
  float*    sAlpha= sMax + 32;
  float*    sV    = (float*)(smem + 16640);        // [128][128] f32 chunk of V

  int tile = blockIdx.x;
  int b = blockIdx.y;
  int wave = threadIdx.x >> 5, lane = threadIdx.x & 31;
  int idx = lane & 15, half = lane >> 4;

  int m0 = tile * 16;
  if (threadIdx.x < 16) { sMax[threadIdx.x] = -INFINITY; sSum[threadIdx.x] = 0.0f; }
  __syncthreads();

  v8f acc1 = {}, acc0 = {};
  const float* qrow = Q + ((size_t)b * S + m0 + idx) * E;
  int e0 = wave * 16;  // output feature tile of this wave
  int jmax = m0 + 15;  // keys valid iff j < query index

  for (int j0 = 0; j0 <= jmax; j0 += 128) {
    // --- stage V[j0:j0+128, :] into LDS (async global->LDS when available)
#if defined(HAVE_ASYNC_LDS)
    {
      const char* gsrc = (const char*)(V + ((size_t)b * S + j0) * E);
      char* ldst = (char*)sV;
#pragma unroll
      for (int i = 0; i < 16; ++i) {
        int e = threadIdx.x + i * 256;  // 4096 x b128
        __builtin_amdgcn_global_load_async_to_lds_b128(
            (i4v*)(gsrc + (size_t)e * 16), (i4v*)(ldst + e * 16), 0, 0);
      }
    }
#else
    for (int i = threadIdx.x; i < 128 * 128; i += 256)
      sV[i] = V[((size_t)b * S + j0) * E + i];
#endif
    // --- prefetch next K chunk while this chunk computes
    if (j0 + 128 <= jmax) {
      const char* kp = (const char*)(Km + ((size_t)b * S + j0 + 128 + (threadIdx.x >> 1)) * E +
                                     (threadIdx.x & 1) * 64);
      __builtin_prefetch(kp, 0, 3);
    }

    // --- scores: wave computes a 16x16 tile at key columns [j0+wave*16, +16)
    int jb = j0 + wave * 16;
    const float* krow = Km + ((size_t)b * S + jb + idx) * E;
    v8f s = {};
    for (int k0 = 0; k0 < E; k0 += 32) {
      v16h af = load_a_frag(qrow, k0, half);
      v16h bf = load_b_frag(krow, k0, half);  // B[k][n] = K[jb+n][k]
      s = WMMA_F16(af, bf, s);
    }
    int j = jb + idx;
#pragma unroll
    for (int r = 0; r < 8; ++r) {
      int qi = m0 + half * 8 + r;
      sS[(half * 8 + r) * 128 + wave * 16 + idx] = (j < qi) ? s[r] * 0.125f : -INFINITY;
    }
    wait_async0();      // own async V loads complete
    __syncthreads();    // publish sS and sV to all waves

    // --- online softmax update: one thread per query row
    if (threadIdx.x < 16) {
      int r = threadIdx.x;
      float mold = sMax[r];
      float mc = -INFINITY;
      for (int jj = 0; jj < 128; ++jj) mc = fmaxf(mc, sS[r * 128 + jj]);
      float mnew = fmaxf(mold, mc);
      float alpha = (mnew == -INFINITY) ? 1.0f : __expf(mold - mnew);
      float lsum = 0.0f;
      for (int jj = 0; jj < 128; ++jj) {
        float p = (mnew == -INFINITY) ? 0.0f : __expf(sS[r * 128 + jj] - mnew);
        lsum += p;
        float qf = (float)qa[b * S + j0 + jj];
        sP1[r * 128 + jj] = (_Float16)(p * qf);
        sP0[r * 128 + jj] = (_Float16)(p * (1.0f - qf));
      }
      sSum[r] = sSum[r] * alpha + lsum;
      sMax[r] = mnew;
      sAlpha[r] = alpha;
    }
    __syncthreads();

    // --- rescale accumulators, then acc += P @ V_chunk (two gated variants)
#pragma unroll
    for (int r = 0; r < 8; ++r) {
      float al = sAlpha[half * 8 + r];
      acc1[r] *= al; acc0[r] *= al;
    }
    for (int k0 = 0; k0 < 128; k0 += 32) {
      v16h a1, a0, bf;
#pragma unroll
      for (int i = 0; i < 8; ++i) {
        a1[i]     = sP1[idx * 128 + k0 + half * 8 + i];
        a1[8 + i] = sP1[idx * 128 + k0 + 16 + half * 8 + i];
        a0[i]     = sP0[idx * 128 + k0 + half * 8 + i];
        a0[8 + i] = sP0[idx * 128 + k0 + 16 + half * 8 + i];
      }
#pragma unroll
      for (int i = 0; i < 16; ++i)
        bf[i] = (_Float16)sV[(k0 + half * 8 + i + (i >= 8 ? 8 : 0)) * 128 + e0 + idx];
      acc1 = WMMA_F16(a1, bf, acc1);
      acc0 = WMMA_F16(a0, bf, acc0);
    }
    __syncthreads();  // before sS/sP/sV are overwritten next chunk
  }

  // --- epilogue: write all four 128-wide slices of p_input
#pragma unroll
  for (int r = 0; r < 8; ++r) {
    int row = half * 8 + r;
    int grow = m0 + row;
    float l = sSum[row];
    float inv = (l > 0.0f) ? 1.0f / l : 0.0f;
    size_t base = ((size_t)b * S + grow) * 512;
    float vv = V[((size_t)b * S + grow) * E + e0 + idx];
    float qf = (float)qa[b * S + grow];
    PIN[base + e0 + idx]        = qf * vv;
    PIN[base + 128 + e0 + idx]  = (1.0f - qf) * vv;
    PIN[base + 256 + e0 + idx]  = acc1[r] * inv;
    PIN[base + 384 + e0 + idx]  = acc0[r] * inv;
  }
}

// ---------------------------------------------------------------------------
// Persistent fused LSTM scan. One block per 16 batch rows; Whh as f16 in LDS
// (128KB), h/c state in LDS, xg[t] double-buffered in LDS via the Tensor Data
// Mover one step ahead (TENSORcnt). 32 waves of WMMA per timestep, then fused
// gate nonlinearities + state update. XG already contains x@Wih^T + bih + bhh.
// ---------------------------------------------------------------------------
#define SCAN_SW_OFF   0
#define SCAN_SH_OFF   (512 * 128 * 2)
#define SCAN_SG_OFF   (SCAN_SH_OFF + 16 * 128 * 2)
#define SCAN_SC_OFF   (SCAN_SG_OFF + 16 * 512 * 4)
#define SCAN_XG_OFF   (SCAN_SC_OFF + 16 * 128 * 4)
#if defined(HAVE_TDM)
#define SCAN_SMEM_BYTES (SCAN_XG_OFF + 2 * 16 * 512 * 4)
#else
#define SCAN_SMEM_BYTES SCAN_XG_OFF
#endif

#if defined(HAVE_TDM)
// Issue a TDM 2D tile load: XG[(b0..b0+15), t, 0..511] (fp32) -> LDS buffer.
static __device__ __forceinline__ void tdm_load_xg(const float* XG, int b0, int t,
                                                   void* lds_dst) {
  const int S = 1024, G = 512;
  unsigned long long ga =
      (unsigned long long)(size_t)(const void*)&XG[((size_t)b0 * S + t) * G];
  unsigned ldsaddr = (unsigned)(size_t)lds_dst;
  tdm_v4u g0;
  g0[0] = 1u;                                   // count=1, user descriptor
  g0[1] = ldsaddr;                              // lds_addr
  g0[2] = (unsigned)ga;                         // global_addr[31:0]
  g0[3] = ((unsigned)(ga >> 32) & 0x01FFFFFFu)  // global_addr[56:32]
          | 0x80000000u;                        // type=2 ("image")
  tdm_v8i g1;
  g1[0] = 0x20000;        // data_size=4B, no multicast
  g1[1] = 0x02000000;     // tensor_dim0 = 512 (bits 79:48 -> low half here)
  g1[2] = 0x00100000;     // tensor_dim0 hi = 0, tensor_dim1 = 16
  g1[3] = 0x02000000;     // tensor_dim1 hi = 0, tile_dim0 = 512
  g1[4] = 16;             // tile_dim1 = 16, tile_dim2 = 0
  g1[5] = 0x80000;        // tensor_dim0_stride = S*G = 524288 (lo)
  g1[6] = 0;              // stride hi, dim1_stride lo
  g1[7] = 0;
  tdm_v4i z4 = {0, 0, 0, 0};
  tdm_v8i z8 = {0, 0, 0, 0, 0, 0, 0, 0};
  __builtin_amdgcn_tensor_load_to_lds(g0, g1, z4, z4, z8, 0);
}
#endif

__global__ __launch_bounds__(1024, 1) void lstm_scan(
    const float* __restrict__ XG, const float* __restrict__ Whh,
    float* __restrict__ Hout) {
  const int S = 1024, Hn = 128, G = 512;
  extern __shared__ char smem[];
  _Float16* sW = (_Float16*)(smem + SCAN_SW_OFF);  // 512*128 f16
  _Float16* sH = (_Float16*)(smem + SCAN_SH_OFF);  // 16*128 f16
  float*    sG = (float*)(smem + SCAN_SG_OFF);     // 16*512 f32
  float*    sC = (float*)(smem + SCAN_SC_OFF);     // 16*128 f32
#if defined(HAVE_TDM)
  float*    sXGbase = (float*)(smem + SCAN_XG_OFF);  // 2 x 16*512 f32
#endif

  int b0 = blockIdx.x * 16;
  int tid = threadIdx.x;

#if defined(HAVE_TDM)
  if (tid < 32) tdm_load_xg(XG, b0, 0, sXGbase);  // one wave issues the TDM op
#endif
  for (int i = tid; i < 512 * 128; i += 1024) sW[i] = (_Float16)Whh[i];
  for (int i = tid; i < 16 * 128; i += 1024) { sH[i] = (_Float16)0.0f; sC[i] = 0.0f; }
  __syncthreads();

  int wave = tid >> 5, lane = tid & 31;
  int idx = lane & 15, half = lane >> 4;
  int n0 = wave * 16;
  int gate = n0 >> 7;  // 0=i,1=f,2=g,3=o

  for (int t = 0; t < S; ++t) {
#if defined(HAVE_TDM)
    int cur = t & 1;
    float* sXGcur = sXGbase + cur * (16 * 512);
    float* sXGnxt = sXGbase + (1 - cur) * (16 * 512);
    if (tid < 32) __builtin_amdgcn_s_wait_tensorcnt(0);  // sXGcur ready
    __syncthreads();                                     // publish to all waves
    if (tid < 32 && t + 1 < S) tdm_load_xg(XG, b0, t + 1, sXGnxt);
#endif
    v8f acc = {};
    for (int k0 = 0; k0 < Hn; k0 += 32) {
      v16h af, bf;
#pragma unroll
      for (int i = 0; i < 8; ++i) {
        af[i]     = sH[idx * 128 + k0 + half * 8 + i];
        af[8 + i] = sH[idx * 128 + k0 + 16 + half * 8 + i];
      }
#pragma unroll
      for (int i = 0; i < 16; ++i)
        bf[i] = sW[(n0 + idx) * 128 + k0 + half * 8 + i + (i >= 8 ? 8 : 0)];
      acc = WMMA_F16(af, bf, acc);
    }
    int col = n0 + idx;
#pragma unroll
    for (int r = 0; r < 8; ++r) {
      int m = half * 8 + r;
#if defined(HAVE_TDM)
      float g = acc[r] + sXGcur[m * G + col];
#else
      float g = acc[r] + XG[((size_t)(b0 + m) * S + t) * G + col];
#endif
      g = (gate == 2) ? tanhf(g) : sigmoidf_(g);
      sG[m * G + col] = g;
    }
    __syncthreads();
    // fused cell/state update: 16*128 elements over 1024 threads
    for (int e = tid; e < 16 * 128; e += 1024) {
      int m = e >> 7, jcol = e & 127;
      float iv = sG[m * G + jcol];
      float fv = sG[m * G + 128 + jcol];
      float gv = sG[m * G + 256 + jcol];
      float ov = sG[m * G + 384 + jcol];
      float c = fv * sC[e] + iv * gv;
      float h = ov * tanhf(c);
      sC[e] = c;
      sH[e] = (_Float16)h;
      Hout[((size_t)(b0 + m) * S + t) * Hn + jcol] = h;
    }
    __syncthreads();
  }
}

// ---------------------------------------------------------------------------
// Glue kernels
// ---------------------------------------------------------------------------
// pv[m,384] = [ v[b, t+1, :128] | p_input[b, t+1, 256:512] ],  m = b*1023 + t
__global__ void build_pv(const float* __restrict__ V, const float* __restrict__ PIN,
                         float* __restrict__ PV, int total) {
  int i = blockIdx.x * blockDim.x + threadIdx.x;
  if (i >= total) return;
  int m = i / 384, k = i % 384;
  int b = m / 1023, t = m % 1023;
  size_t row = (size_t)b * 1024 + t + 1;
  PV[i] = (k < 128) ? V[row * 128 + k] : PIN[row * 512 + 256 + (k - 128)];
}

// pred[m, 0:128] = H1[b*1024 + t, :]  (xm GEMM fills cols 128:256)
__global__ void build_pred_left(const float* __restrict__ H1, float* __restrict__ PRED,
                                int total) {
  int i = blockIdx.x * blockDim.x + threadIdx.x;
  if (i >= total) return;
  int m = i >> 7, jcol = i & 127;
  int b = m / 1023, t = m % 1023;
  PRED[(size_t)m * 256 + jcol] = H1[((size_t)b * 1024 + t) * 128 + jcol];
}

// Final head: logits = y1 @ W2^T + b2; BCE + probs + masked targets.
__global__ __launch_bounds__(256) void head_kernel(
    const float* __restrict__ Y1, const float* __restrict__ W2,
    const float* __restrict__ b2, const float* __restrict__ target,
    float* __restrict__ outProbs, float* __restrict__ outTgt,
    float* __restrict__ red, int M) {
  int wave = threadIdx.x >> 5, lane = threadIdx.x & 31;
  int m = blockIdx.x * 8 + wave;
  if (m >= M) return;
  float p = 0.0f;
  for (int jcol = lane; jcol < 128; jcol += 32) p += Y1[(size_t)m * 128 + jcol] * W2[jcol];
#pragma unroll
  for (int o = 16; o > 0; o >>= 1) p += __shfl_xor(p, o, 32);
  if (lane == 0) {
    float z = p + b2[0];
    float t = target[m];
    float mask = (t >= 0.0f) ? 1.0f : 0.0f;
    float bce = fmaxf(z, 0.0f) - z * t + log1pf(__expf(-fabsf(z)));
    outProbs[m] = sigmoidf_(z) * mask;
    outTgt[m] = t * mask;
    atomicAdd(&red[0], bce * mask);
    atomicAdd(&red[1], mask);
  }
}

__global__ void zero_red(float* red) {
  if (threadIdx.x < 2) red[threadIdx.x] = 0.0f;
}
__global__ void finalize_loss(const float* __restrict__ red, float* __restrict__ out) {
  out[0] = red[0] / red[1];
}

// ---------------------------------------------------------------------------
extern "C" void kernel_launch(void* const* d_in, const int* in_sizes, int n_in,
                              void* d_out, int out_size, void* d_ws, size_t ws_size,
                              hipStream_t stream) {
  const int B = 32, S = 1024;
  const int MS = B * S;          // 32768
  const int MT = B * (S - 1);    // 32736

  // Inputs (setup_inputs dict order)
  const int*   qa    = (const int*)d_in[1];
  const int*   pid   = (const int*)d_in[2];
  const float* tgt   = (const float*)d_in[3];
  const float* emb   = (const float*)d_in[4];
  const float* Wq    = (const float*)d_in[5];
  const float* bq    = (const float*)d_in[6];
  const float* Wk    = (const float*)d_in[7];
  const float* bk    = (const float*)d_in[8];
  const float* Wv    = (const float*)d_in[9];
  const float* bv    = (const float*)d_in[10];
  const float* Wy    = (const float*)d_in[11];
  const float* by    = (const float*)d_in[12];
  const float* Wx    = (const float*)d_in[13];
  const float* bx    = (const float*)d_in[14];
  const float* W1    = (const float*)d_in[15];
  const float* b1    = (const float*)d_in[16];
  const float* W2    = (const float*)d_in[17];
  const float* b2    = (const float*)d_in[18];
  const float* Wih0  = (const float*)d_in[19];
  const float* Whh0  = (const float*)d_in[20];
  const float* bih0  = (const float*)d_in[21];
  const float* bhh0  = (const float*)d_in[22];
  const float* Wih1  = (const float*)d_in[23];
  const float* Whh1  = (const float*)d_in[24];
  const float* bih1  = (const float*)d_in[25];
  const float* bhh1  = (const float*)d_in[26];

  // Workspace layout (bytes)
  char* ws = (char*)d_ws;
  size_t off = 0;
  auto alloc = [&](size_t bytes) { size_t o = off; off += (bytes + 255) & ~(size_t)255; return o; };
  float* X    = (float*)(ws + alloc((size_t)MS * 64 * 4));
  float* Qb   = (float*)(ws + alloc((size_t)MS * 128 * 4));
  float* Kb   = (float*)(ws + alloc((size_t)MS * 128 * 4));
  float* Vb   = (float*)(ws + alloc((size_t)MS * 128 * 4));
  float* PIN  = (float*)(ws + alloc((size_t)MS * 512 * 4));
  float* Yb   = (float*)(ws + alloc((size_t)MS * 128 * 4));
  float* XG   = (float*)(ws + alloc((size_t)MS * 512 * 4));   // reused by both LSTM layers
  float* H0   = (float*)(ws + alloc((size_t)MS * 128 * 4));
  float* H1   = (float*)(ws + alloc((size_t)MS * 128 * 4));
  float* PV   = (float*)(ws + alloc((size_t)MT * 384 * 4));
  float* PRED = (float*)(ws + alloc((size_t)MT * 256 * 4));
  float* Y1b  = (float*)(ws + alloc((size_t)MT * 128 * 4));
  float* RED  = (float*)(ws + alloc(2 * 4));
  (void)ws_size; (void)n_in; (void)in_sizes; (void)out_size;

  float* out = (float*)d_out;
  float* outProbs = out + 1;
  float* outTgt = out + 1 + MT;

  (void)hipFuncSetAttribute(reinterpret_cast<const void*>(attn_kernel),
                            hipFuncAttributeMaxDynamicSharedMemorySize, ATTN_SMEM_BYTES);
  (void)hipFuncSetAttribute(reinterpret_cast<const void*>(lstm_scan),
                            hipFuncAttributeMaxDynamicSharedMemorySize, SCAN_SMEM_BYTES);

  zero_red<<<1, 32, 0, stream>>>(RED);

  // 1) x = emb[pid]
  gather_x<<<(MS * 64 + 255) / 256, 256, 0, stream>>>(pid, emb, X, MS * 64);

  // 2) Q/K/V projections (M=32768, K=64, N=128)
  gemm_wmma<<<dim3(MS / 16, 1), 256, 0, stream>>>(X, 64, Wq, 64, bq, nullptr, Qb, 128, 0, MS, 128, 64, 0);
  gemm_wmma<<<dim3(MS / 16, 1), 256, 0, stream>>>(X, 64, Wk, 64, bk, nullptr, Kb, 128, 0, MS, 128, 64, 0);
  gemm_wmma<<<dim3(MS / 16, 1), 256, 0, stream>>>(X, 64, Wv, 64, bv, nullptr, Vb, 128, 0, MS, 128, 64, 0);

  // 3) fused causal attention -> p_input[MS,512]
  attn_kernel<<<dim3(S / 16, B), 256, ATTN_SMEM_BYTES, stream>>>(Qb, Kb, Vb, qa, PIN);

  // 4) y = p_input @ Wy^T + by  (K=512, N=128)
  gemm_wmma<<<dim3(MS / 16, 1), 256, 0, stream>>>(PIN, 512, Wy, 512, by, nullptr, Yb, 128, 0, MS, 128, 512, 0);

  // 5) LSTM layer 0: xg = y @ Wih0^T + (bih0 + bhh0); scan
  gemm_wmma<<<dim3(MS / 16, 4), 256, 0, stream>>>(Yb, 128, Wih0, 128, bih0, bhh0, XG, 512, 0, MS, 512, 128, 0);
  lstm_scan<<<2, 1024, SCAN_SMEM_BYTES, stream>>>(XG, Whh0, H0);

  // 6) LSTM layer 1
  gemm_wmma<<<dim3(MS / 16, 4), 256, 0, stream>>>(H0, 128, Wih1, 128, bih1, bhh1, XG, 512, 0, MS, 512, 128, 0);
  lstm_scan<<<2, 1024, SCAN_SMEM_BYTES, stream>>>(XG, Whh1, H1);

  // 7) pv = [v[:,1:], p_dist[:,1:]]; xm = pv @ Wx^T + bx -> pred[:,128:256]
  build_pv<<<(MT * 384 + 255) / 256, 256, 0, stream>>>(Vb, PIN, PV, MT * 384);
  gemm_wmma<<<dim3(MT / 16, 1), 256, 0, stream>>>(PV, 384, Wx, 384, bx, nullptr, PRED, 256, 128, MT, 128, 384, 0);
  build_pred_left<<<(MT * 128 + 255) / 256, 256, 0, stream>>>(H1, PRED, MT * 128);

  // 8) y1 = relu(pred @ W1^T + b1)  (K=256, N=128)
  gemm_wmma<<<dim3(MT / 16, 1), 256, 0, stream>>>(PRED, 256, W1, 256, b1, nullptr, Y1b, 128, 0, MT, 128, 256, 1);

  // 9) head: logits, BCE reduction, probs, masked targets
  head_kernel<<<(MT + 7) / 8, 256, 0, stream>>>(Y1b, W2, b2, tgt, outProbs, outTgt, RED, MT);
  finalize_loss<<<1, 1, 0, stream>>>(RED, out);
}
